// APPNPNet_79156247266009
// MI455X (gfx1250) — matrix operand order repaired and checked by reading, not compiled, not run on hardware
//
#include <hip/hip_runtime.h>
#include <stdint.h>

#define IN_C  512
#define HID_C 128
#define OUT_C 40
#define K_ITERS 5
// ALPHA = 0.5, (1-ALPHA) = 0.5

typedef __attribute__((ext_vector_type(16))) __bf16 v16bf;
typedef __attribute__((ext_vector_type(8)))  float  v8f;

// round-to-nearest-even f32 -> bf16 (used only in one-shot pack kernels)
__device__ __forceinline__ __bf16 f2bf_rne(float f) {
  union { float f; unsigned u; } in; in.f = f;
  unsigned r = (in.u + 0x7FFFu + ((in.u >> 16) & 1u)) >> 16;
  union { unsigned short s; __bf16 b; } out; out.s = (unsigned short)r;
  return out.b;
}

// v_perm_b32: pack two f32 into one dword of truncated bf16 (lo in [15:0], hi in [31:16])
__device__ __forceinline__ unsigned pack2bf(float lo, float hi) {
  return __builtin_amdgcn_perm(__float_as_uint(hi), __float_as_uint(lo), 0x07060302u);
}

// ---------------- degree / normalization ----------------

__global__ void degree_count(const long long* __restrict__ col, unsigned* __restrict__ deg,
                             long long E) {
  long long e = (long long)blockIdx.x * blockDim.x + threadIdx.x;
  if (e < E) atomicAdd(&deg[(int)col[e]], 1u);
}

__global__ void compute_dinv(const unsigned* __restrict__ deg, float* __restrict__ dinv, int N) {
  int i = blockIdx.x * blockDim.x + threadIdx.x;
  if (i < N) {
    float d = (float)deg[i] + 1.0f;   // +1 self loop -> always > 0
    dinv[i] = rsqrtf(d);
  }
}

// ---------------- one-shot bf16 fragment packing ----------------
// WMMA bf16 fragment layouts (ISA 7.12.2, wave32):
//  A (16x32): lane L holds row m=L%16, elems i: k = (L>=16?8:0) + (i<8? i : i+8)
//  B (32x16): lane L holds col n=L%16, elems i: k = (L>=16?8:0) + (i<8? i : i+8)

// w1p layout: [chunk c:16][tile t:8][lane:32][16 bf16]  (chunk-major: one k-chunk = 8KB contiguous)
__global__ void pack_w1(const float* __restrict__ W1, __bf16* __restrict__ w1p) {
  int idx = blockIdx.x * 256 + threadIdx.x;        // (c,t,lane)
  if (idx >= 16 * 8 * 32) return;
  int lane = idx & 31, t = (idx >> 5) & 7, c = idx >> 8;
  int n = (t << 4) + (lane & 15);
  int kbase = c * 32 + ((lane & 16) ? 8 : 0);
  __bf16* dst = w1p + (size_t)idx * 16;
#pragma unroll
  for (int i = 0; i < 16; ++i)
    dst[i] = f2bf_rne(W1[(size_t)(kbase + (i < 8 ? i : i + 8)) * HID_C + n]);
}

// w2p layout: [t:3][c:4][lane:32][16 bf16], zero padded for n >= OUT_C
__global__ void pack_w2(const float* __restrict__ W2, __bf16* __restrict__ w2p) {
  int idx = blockIdx.x * 256 + threadIdx.x;        // (t,c,lane)
  if (idx >= 3 * 4 * 32) return;
  int lane = idx & 31, c = (idx >> 5) & 3, t = idx >> 7;
  int n = (t << 4) + (lane & 15);
  int kbase = c * 32 + ((lane & 16) ? 8 : 0);
  __bf16* dst = w2p + (size_t)idx * 16;
#pragma unroll
  for (int i = 0; i < 16; ++i)
    dst[i] = (n < OUT_C) ? f2bf_rne(W2[(size_t)(kbase + (i < 8 ? i : i + 8)) * OUT_C + n]) : f2bf_rne(0.0f);
}

// xp layout: [tile][chunk c:16][lane:32][16 bf16]  (A fragments, one per (tile, k-chunk))
__global__ void pack_x(const float* __restrict__ x, __bf16* __restrict__ xp,
                       int nTiles, int N) {
  long long idx = (long long)blockIdx.x * 256 + threadIdx.x;   // (tile,c,lane)
  int lane = (int)(idx & 31);
  int c = (int)((idx >> 5) & 15);
  int tile = (int)(idx >> 9);
  if (tile >= nTiles) return;
  int m = tile * 16 + (lane & 15);
  if (m >= N) m = N - 1;
  const float* src = x + (size_t)m * IN_C + c * 32 + ((lane & 16) ? 8 : 0);
  __bf16* dst = xp + (size_t)idx * 16;
#pragma unroll
  for (int i = 0; i < 16; ++i)
    dst[i] = f2bf_rne(src[i < 8 ? i : i + 8]);
}

// ---------------- GEMM1: h0 = x @ W1 + b1 ----------------
// 8 waves/block, one 16x128 strip per wave. Double-buffered LDS pipeline:
// while computing k-chunk c from one 8KB buffer, chunk c+1 streams into the
// other buffer via global_load_async_to_lds_b128; s_wait_asynccnt 0x2 waits
// only for the older pair (ASYNCcnt completes in order), overlapping DMA with WMMA.

__global__ void __launch_bounds__(256) gemm1_wmma(const __bf16* __restrict__ xp,
                                                  const __bf16* __restrict__ w1p,
                                                  const float* __restrict__ b1,
                                                  float* __restrict__ h0, int nTiles) {
  __shared__ __align__(32) __bf16 ldsB[2][8 * 32 * 16];   // 2 x 8KB B-fragment buffers
  const int tid  = threadIdx.x;
  const int wave = tid >> 5;
  const int lane = tid & 31;
  const int tile = blockIdx.x * 8 + wave;
  // wave-uniform predicate routed through an SGPR -> pure scalar branch,
  // EXEC stays all-ones at every WMMA
  const int active = __builtin_amdgcn_readfirstlane(tile < nTiles ? 1 : 0);
  const int mlane = lane & 15;
  const int mbase = (lane & 16) ? 8 : 0;

  v8f acc[8];
#pragma unroll
  for (int t = 0; t < 8; ++t)
#pragma unroll
    for (int r = 0; r < 8; ++r) acc[t][r] = 0.0f;

  const __bf16* aSrc = xp + ((size_t)tile * 16 * 32 + (size_t)lane) * 16;
  const unsigned ldsAddr[2] = {
    (unsigned)(uintptr_t)&ldsB[0][0] + (unsigned)tid * 32u,
    (unsigned)(uintptr_t)&ldsB[1][0] + (unsigned)tid * 32u
  };

  // prologue: stage chunk 0 into buffer 0 (256 lanes x 32B async copies)
  {
    unsigned long long g =
        (unsigned long long)(uintptr_t)((const char*)w1p + (size_t)tid * 32);
    asm volatile("global_load_async_to_lds_b128 %0, %1, off"
                 :: "v"(ldsAddr[0]), "v"(g) : "memory");
    asm volatile("global_load_async_to_lds_b128 %0, %1, off offset:16"
                 :: "v"(ldsAddr[0]), "v"(g) : "memory");
  }

  for (int c = 0; c < 16; ++c) {
    if (c + 1 < 16) {
      // kick off DMA for the next chunk into the other buffer
      unsigned long long g = (unsigned long long)(uintptr_t)
          ((const char*)w1p + (size_t)(c + 1) * 8192 + (size_t)tid * 32);
      asm volatile("global_load_async_to_lds_b128 %0, %1, off"
                   :: "v"(ldsAddr[(c + 1) & 1]), "v"(g) : "memory");
      asm volatile("global_load_async_to_lds_b128 %0, %1, off offset:16"
                   :: "v"(ldsAddr[(c + 1) & 1]), "v"(g) : "memory");
      asm volatile("s_wait_asynccnt 0x2" ::: "memory");  // older pair (chunk c) done
    } else {
      asm volatile("s_wait_asynccnt 0x0" ::: "memory");
    }
    __syncthreads();   // chunk c visible from every wave's copies

    if (active) {
      v16bf a = *(const v16bf*)aSrc;     // 32B contiguous A fragment (b128 x2)
      aSrc += 32 * 16;
      const __bf16* bBuf = &ldsB[c & 1][0];
#pragma unroll
      for (int t = 0; t < 8; ++t) {
        v16bf b = *(const v16bf*)(bBuf + (size_t)((t << 5) + lane) * 16);  // ds_load_b128 x2
        acc[t] = __builtin_amdgcn_wmma_f32_16x16x32_bf16(false, a, false, b,
                                                         (short)0, acc[t], false, false);
      }
    }
    __syncthreads();   // all reads of buf[c&1] done before iter c+1 overwrites it
  }

  if (active) {
    const int row0 = tile << 4;
#pragma unroll
    for (int t = 0; t < 8; ++t) {
      const int n = (t << 4) + mlane;
      const float bias = b1[n];
#pragma unroll
      for (int r = 0; r < 8; ++r)
        h0[(size_t)(row0 + mbase + r) * HID_C + n] = acc[t][r] + bias;
    }
  }
}

// ---------------- propagation ----------------

__global__ void prop_init(const float* __restrict__ x0, const float* __restrict__ h,
                          const float* __restrict__ dinv, float* __restrict__ hn,
                          long long nVec) {
  long long i = (long long)blockIdx.x * blockDim.x + threadIdx.x; // float4 index
  if (i >= nVec) return;
  int node = (int)(i >> 5);              // 32 float4 per node (HID_C/4)
  float di = dinv[node];
  float w  = 0.5f * di * di;
  float4 a = ((const float4*)x0)[i];
  float4 b = ((const float4*)h)[i];
  float4 o;
  o.x = 0.5f * a.x + w * b.x;
  o.y = 0.5f * a.y + w * b.y;
  o.z = 0.5f * a.z + w * b.z;
  o.w = 0.5f * a.w + w * b.w;
  ((float4*)hn)[i] = o;
}

__global__ void edge_prop(const long long* __restrict__ row, const long long* __restrict__ col,
                          const float* __restrict__ dinv, const float* __restrict__ h,
                          float* __restrict__ hn, long long E) {
  const int lane = threadIdx.x & 31;
  const long long e = (long long)blockIdx.x * 8 + (threadIdx.x >> 5);
  if (e >= E) return;
  const int r = (int)row[e];
  const int c = (int)col[e];
  if (e + 8 < E) {                       // global_prefetch_b8 of an upcoming source row
    const int rn = (int)row[e + 8];
    __builtin_prefetch(h + (size_t)rn * HID_C + lane * 4, 0, 1);
  }
  const float w = 0.5f * dinv[r] * dinv[c];
  const float4 v = ((const float4*)(h + (size_t)r * HID_C))[lane];
  float* dst = hn + (size_t)c * HID_C + (size_t)lane * 4;
  __hip_atomic_fetch_add(dst + 0, w * v.x, __ATOMIC_RELAXED, __HIP_MEMORY_SCOPE_AGENT);
  __hip_atomic_fetch_add(dst + 1, w * v.y, __ATOMIC_RELAXED, __HIP_MEMORY_SCOPE_AGENT);
  __hip_atomic_fetch_add(dst + 2, w * v.z, __ATOMIC_RELAXED, __HIP_MEMORY_SCOPE_AGENT);
  __hip_atomic_fetch_add(dst + 3, w * v.w, __ATOMIC_RELAXED, __HIP_MEMORY_SCOPE_AGENT);
}

// ---------------- GEMM2: out = h @ W2 + b2 (OUT_C padded to 3x16) ----------------

__global__ void gemm2_wmma(const float* __restrict__ h, const __bf16* __restrict__ w2p,
                           const float* __restrict__ b2, float* __restrict__ out, int nTiles) {
  const int wave = threadIdx.x >> 5;
  const int lane = threadIdx.x & 31;
  const int tile = blockIdx.x * 8 + wave;
  if (__builtin_amdgcn_readfirstlane(tile >= nTiles ? 1 : 0)) return;  // scalar branch
  const int row0  = tile << 4;
  const int mlane = lane & 15;
  const int abase = (lane & 16) ? 8 : 0;
  const int mbase = (lane & 16) ? 8 : 0;

  v8f acc[3];
#pragma unroll
  for (int t = 0; t < 3; ++t)
#pragma unroll
    for (int r = 0; r < 8; ++r) acc[t][r] = 0.0f;

  const float* hr = h + (size_t)(row0 + mlane) * HID_C;
#pragma unroll
  for (int c = 0; c < 4; ++c) {
    const float4* p = (const float4*)(hr + c * 32 + abase);
    float4 q0 = p[0], q1 = p[1];         // k lo run (8 floats)
    float4 q2 = p[4], q3 = p[5];         // k hi run (+16 floats)
    union { unsigned u[8]; v16bf v; } av;
    av.u[0] = pack2bf(q0.x, q0.y); av.u[1] = pack2bf(q0.z, q0.w);
    av.u[2] = pack2bf(q1.x, q1.y); av.u[3] = pack2bf(q1.z, q1.w);
    av.u[4] = pack2bf(q2.x, q2.y); av.u[5] = pack2bf(q2.z, q2.w);
    av.u[6] = pack2bf(q3.x, q3.y); av.u[7] = pack2bf(q3.z, q3.w);
#pragma unroll
    for (int t = 0; t < 3; ++t) {
      v16bf b = *(const v16bf*)(w2p + (size_t)(((t << 2) + c) * 32 + lane) * 16);
      acc[t] = __builtin_amdgcn_wmma_f32_16x16x32_bf16(false, av.v, false, b,
                                                       (short)0, acc[t], false, false);
    }
  }

#pragma unroll
  for (int t = 0; t < 3; ++t) {
    const int n = (t << 4) + mlane;
    if (n < OUT_C) {
      const float bias = b2[n];
#pragma unroll
      for (int r = 0; r < 8; ++r)
        out[(size_t)(row0 + mbase + r) * OUT_C + n] = acc[t][r] + bias;
    }
  }
}

// ---------------- host ----------------

extern "C" void kernel_launch(void* const* d_in, const int* in_sizes, int n_in,
                              void* d_out, int out_size, void* d_ws, size_t ws_size,
                              hipStream_t stream) {
  const float*     x  = (const float*)d_in[0];
  const long long* ei = (const long long*)d_in[1];   // int64 per reference: [2, E]
  const float*     W1 = (const float*)d_in[2];
  const float*     b1 = (const float*)d_in[3];
  const float*     W2 = (const float*)d_in[4];
  const float*     b2 = (const float*)d_in[5];
  float* out = (float*)d_out;

  const int       N = in_sizes[0] / IN_C;
  const long long E = (long long)in_sizes[1] / 2;
  const long long* rowIdx = ei;
  const long long* colIdx = ei + E;
  const int nTiles = (N + 15) / 16;

  // workspace: x0 | hA | hB | xp(bf16) | w1p | w2p | deg | dinv
  const size_t featBytes = (size_t)N * HID_C * sizeof(float);
  char* ws = (char*)d_ws;
  float*    x0  = (float*)ws;    ws += featBytes;
  float*    hA  = (float*)ws;    ws += featBytes;
  float*    hB  = (float*)ws;    ws += featBytes;
  __bf16*   xp  = (__bf16*)ws;   ws += (size_t)nTiles * 16 * 32 * 16 * sizeof(__bf16);
  __bf16*   w1p = (__bf16*)ws;   ws += (size_t)16 * 8 * 32 * 16 * sizeof(__bf16);
  __bf16*   w2p = (__bf16*)ws;   ws += (size_t)3 * 4 * 32 * 16 * sizeof(__bf16);
  unsigned* deg = (unsigned*)ws; ws += (size_t)N * sizeof(unsigned);
  float*    dinv = (float*)ws;

  // normalization
  hipMemsetAsync(deg, 0, (size_t)N * sizeof(unsigned), stream);
  degree_count<<<dim3((unsigned)((E + 255) / 256)), dim3(256), 0, stream>>>(colIdx, deg, E);
  compute_dinv<<<dim3((N + 255) / 256), dim3(256), 0, stream>>>(deg, dinv, N);

  // one-shot bf16 fragment packing
  pack_w1<<<dim3((16 * 8 * 32 + 255) / 256), dim3(256), 0, stream>>>(W1, w1p);
  pack_w2<<<dim3((3 * 4 * 32 + 255) / 256), dim3(256), 0, stream>>>(W2, w2p);
  {
    long long tot = (long long)nTiles * 16 * 32;
    pack_x<<<dim3((unsigned)((tot + 255) / 256)), dim3(256), 0, stream>>>(x, xp, nTiles, N);
  }

  // GEMM1 -> x0 (kept intact as the APPNP teleport term)
  gemm1_wmma<<<dim3((nTiles + 7) / 8), dim3(256), 0, stream>>>(xp, w1p, b1, x0, nTiles);

  // K propagation rounds, 3-buffer ping-pong (x0 never overwritten)
  float* cur = x0;
  float* nxt = hA;
  float* spare = hB;
  const long long nVec = (long long)N * (HID_C / 4);
  const long long edgeBlocks = (E + 7) / 8;
  for (int it = 0; it < K_ITERS; ++it) {
    prop_init<<<dim3((unsigned)((nVec + 255) / 256)), dim3(256), 0, stream>>>(x0, cur, dinv, nxt, nVec);
    edge_prop<<<dim3((unsigned)edgeBlocks), dim3(256), 0, stream>>>(rowIdx, colIdx, dinv, cur, nxt, E);
    float* done = nxt;
    nxt = (it == 0) ? spare : cur;
    cur = done;
  }

  // GEMM2 -> out
  gemm2_wmma<<<dim3((nTiles + 7) / 8), dim3(256), 0, stream>>>(cur, w2p, b2, out, nTiles);
}
